// Net_62878321213706
// MI455X (gfx1250) — matrix operand order repaired and checked by reading, not compiled, not run on hardware
//
#include <hip/hip_runtime.h>
#include <math.h>

// ---------------------------------------------------------------------------
// PointNet++ forward for MI455X (gfx1250, wave32, WMMA 16x16x32 f16/f32-acc).
// All GEMMs lowered to v_wmma_f32_16x16x32_f16; SA modules fused per-group in
// LDS so no giant [M*K, F] intermediates hit HBM (compute-bound workload:
// ~35 GFLOP over ~15 MB of live data vs 23.3 TB/s HBM).
// ---------------------------------------------------------------------------

typedef __attribute__((ext_vector_type(16))) _Float16 v16h;
typedef __attribute__((ext_vector_type(8)))  float    v8f;

#define DEV __device__ __forceinline__
#define NEGINF (-3.0e38f)

DEV v8f wmma_f16(v16h a, v16h b, v8f c) {
  // D = A(16x32) * B(32x16) + C, f32 accumulate
  return __builtin_amdgcn_wmma_f32_16x16x32_f16(false, a, false, b, (short)0, c,
                                                false, false);
}

// A fragment: 16x32 f16 tile, row-major with leading dim lda (elements).
// ISA 7.12.2: lane l holds row m=l%16; half h=l/16 selects K-subrange;
// VGPR i<4 -> K = h*8+2i, VGPR i>=4 -> K = 16+h*8+2(i-4).
DEV v16h load_a_frag(const _Float16* A, int lda) {
  const int l = threadIdx.x & 31, m = l & 15, h = l >> 4;
  v16h a;
#pragma unroll
  for (int i = 0; i < 8; ++i) {
    const int kb = (i < 4) ? (h * 8 + 2 * i) : (16 + h * 8 + 2 * (i - 4));
    a[2 * i]     = A[m * lda + kb];
    a[2 * i + 1] = A[m * lda + kb + 1];
  }
  return a;
}

// B fragment from pre-transposed weights WT[n*ldk + k] (i.e. B[k][n]):
// lane l holds column n=l%16; element j<8 -> K=h*8+j, j>=8 -> K=16+h*8+(j-8).
DEV v16h load_b_fragT(const _Float16* WT, int ldk) {
  const int l = threadIdx.x & 31, n = l & 15, h = l >> 4;
  v16h b;
#pragma unroll
  for (int j = 0; j < 16; ++j) b[j] = WT[n * ldk + h * 8 + j + ((j >= 8) ? 8 : 0)];
  return b;
}

// ---------------------------------------------------------------------------
// Weight pack: W[K x N] f32 row-major -> WT[N x Kp] f16 (zero-padded K)
// ---------------------------------------------------------------------------
__global__ __launch_bounds__(256) void k_packWT(const float* __restrict__ W,
                                                _Float16* __restrict__ WT,
                                                int K, int N, int Kp) {
  int t = blockIdx.x * 256 + threadIdx.x;
  if (t >= N * Kp) return;
  int n = t / Kp, k = t % Kp;
  WT[t] = (k < K) ? (_Float16)W[(size_t)k * N + n] : (_Float16)0.0f;
}

// ---------------------------------------------------------------------------
// Curvature: 10-NN covariance, lambda_min / trace  (one cloud per blockIdx.x)
// ---------------------------------------------------------------------------
__global__ __launch_bounds__(256) void k_curv(const float* __restrict__ pos,
                                              float* __restrict__ curv) {
  __shared__ float sx[2048], sy[2048], sz[2048];
  const int b = blockIdx.x;
  const int p = blockIdx.y * 256 + threadIdx.x;
  for (int i = threadIdx.x; i < 2048; i += 256) {
    const float* q = pos + (size_t)(b * 2048 + i) * 3;
    sx[i] = q[0]; sy[i] = q[1]; sz[i] = q[2];
  }
  __syncthreads();
  const float px = sx[p], py = sy[p], pz = sz[p];
  float bd[10]; int bi[10];
#pragma unroll
  for (int j = 0; j < 10; ++j) { bd[j] = 3.0e38f; bi[j] = 0; }
  for (int q = 0; q < 2048; ++q) {
    float dx = sx[q] - px, dy = sy[q] - py, dz = sz[q] - pz;
    float d2 = dx * dx + dy * dy + dz * dz;
    if (d2 < bd[9]) {
      bd[9] = d2; bi[9] = q;
#pragma unroll
      for (int t = 8; t >= 0; --t) {
        if (bd[t] > bd[t + 1]) {
          float td = bd[t]; bd[t] = bd[t + 1]; bd[t + 1] = td;
          int ti = bi[t]; bi[t] = bi[t + 1]; bi[t + 1] = ti;
        }
      }
    }
  }
  float nbx[10], nby[10], nbz[10];
  float mx = 0.f, my = 0.f, mz = 0.f;
#pragma unroll
  for (int j = 0; j < 10; ++j) {
    nbx[j] = sx[bi[j]]; nby[j] = sy[bi[j]]; nbz[j] = sz[bi[j]];
    mx += nbx[j]; my += nby[j]; mz += nbz[j];
  }
  mx *= 0.1f; my *= 0.1f; mz *= 0.1f;
  float cxx = 0, cxy = 0, cxz = 0, cyy = 0, cyz = 0, czz = 0;
#pragma unroll
  for (int j = 0; j < 10; ++j) {
    float dx = nbx[j] - mx, dy = nby[j] - my, dz = nbz[j] - mz;
    cxx += dx * dx; cxy += dx * dy; cxz += dx * dz;
    cyy += dy * dy; cyz += dy * dz; czz += dz * dz;
  }
  cxx *= 0.1f; cxy *= 0.1f; cxz *= 0.1f; cyy *= 0.1f; cyz *= 0.1f; czz *= 0.1f;
  // analytic eigenvalues of symmetric 3x3
  float tr = cxx + cyy + czz;
  float q3 = tr * (1.0f / 3.0f);
  float b00 = cxx - q3, b11 = cyy - q3, b22 = czz - q3;
  float p1 = cxy * cxy + cxz * cxz + cyz * cyz;
  float p2 = b00 * b00 + b11 * b11 + b22 * b22 + 2.0f * p1;
  float pp = sqrtf(p2 * (1.0f / 6.0f) + 1e-30f);
  float inv = 1.0f / pp;
  float det = b00 * (b11 * b22 - cyz * cyz) - cxy * (cxy * b22 - cyz * cxz) +
              cxz * (cxy * cyz - b11 * cxz);
  float r = det * 0.5f * inv * inv * inv;
  r = fminf(1.0f, fmaxf(-1.0f, r));
  float phi = acosf(r) * (1.0f / 3.0f);
  float emin = q3 + 2.0f * pp * cosf(phi + 2.0943951023931953f);  // smallest
  curv[b * 2048 + p] = emin / (tr + 1e-8f);
}

// ---------------------------------------------------------------------------
// Curvature-weighted FPS: one block per cloud, cloud resident in LDS.
// ---------------------------------------------------------------------------
__global__ __launch_bounds__(256) void k_fps(const float* __restrict__ pos,
                                             const float* __restrict__ curv,
                                             int Np, int m,
                                             int* __restrict__ idx_out) {
  __shared__ float sx[2048], sy[2048], sz[2048], smind[2048], sw[2048];
  __shared__ float s_rv[8];
  __shared__ int s_ri[8];
  __shared__ int s_sel;
  const int b = blockIdx.x, tid = threadIdx.x;
  for (int i = tid; i < Np; i += 256) {
    const float* q = pos + (size_t)(b * Np + i) * 3;
    sx[i] = q[0]; sy[i] = q[1]; sz[i] = q[2];
    sw[i] = 1.0f + 10.0f * curv[b * Np + i];
  }
  __syncthreads();
  const float x0 = sx[0], y0 = sy[0], z0 = sz[0];
  for (int i = tid; i < Np; i += 256) {
    float dx = sx[i] - x0, dy = sy[i] - y0, dz = sz[i] - z0;
    smind[i] = dx * dx + dy * dy + dz * dz;
  }
  if (tid == 0) idx_out[b * m] = 0;
  __syncthreads();
  for (int j = 1; j < m; ++j) {
    float best = -1.0f; int bsel = 0x7fffffff;
    for (int i = tid; i < Np; i += 256) {
      float v = smind[i] * sw[i];
      if (v > best || (v == best && i < bsel)) { best = v; bsel = i; }
    }
#pragma unroll
    for (int off = 16; off > 0; off >>= 1) {
      float ov = __shfl_down(best, off, 32);
      int oi = __shfl_down(bsel, off, 32);
      if (ov > best || (ov == best && oi < bsel)) { best = ov; bsel = oi; }
    }
    if ((tid & 31) == 0) { s_rv[tid >> 5] = best; s_ri[tid >> 5] = bsel; }
    __syncthreads();
    if (tid == 0) {
      float bv = s_rv[0]; int bi = s_ri[0];
      for (int t = 1; t < 8; ++t)
        if (s_rv[t] > bv || (s_rv[t] == bv && s_ri[t] < bi)) { bv = s_rv[t]; bi = s_ri[t]; }
      s_sel = bi;
      idx_out[b * m + j] = bi;
    }
    __syncthreads();
    const int sel = s_sel;
    const float qx = sx[sel], qy = sy[sel], qz = sz[sel];
    for (int i = tid; i < Np; i += 256) {
      float dx = sx[i] - qx, dy = sy[i] - qy, dz = sz[i] - qz;
      smind[i] = fminf(smind[i], dx * dx + dy * dy + dz * dz);
    }
    __syncthreads();
  }
}

// ---------------------------------------------------------------------------
// Gather sampled positions (and optionally curvature) by FPS index
// ---------------------------------------------------------------------------
__global__ __launch_bounds__(256) void k_gather(const float* __restrict__ pos,
                                                const float* __restrict__ curv,
                                                const int* __restrict__ idx,
                                                int Nsrc, int m, int total,
                                                float* __restrict__ opos,
                                                float* __restrict__ ocurv) {
  int t = blockIdx.x * 256 + threadIdx.x;
  if (t >= total) return;
  int b = t / m;
  int i = idx[t];
  const float* q = pos + (size_t)(b * Nsrc + i) * 3;
  opos[t * 3 + 0] = q[0]; opos[t * 3 + 1] = q[1]; opos[t * 3 + 2] = q[2];
  if (ocurv) ocurv[t] = curv[b * Nsrc + i];
}

// ---------------------------------------------------------------------------
// Radius grouping: K=64 nearest within radius per centroid (set semantics).
// One centroid per thread, candidate set kept in LDS (replace-current-max).
// ---------------------------------------------------------------------------
__global__ __launch_bounds__(64) void k_group(const float* __restrict__ cpos,
                                              const float* __restrict__ ppos,
                                              int Np, int m, float r2,
                                              int* __restrict__ nidx) {
  __shared__ float sd[64 * 64];
  __shared__ int si[64 * 64];
  const int c = blockIdx.x * 64 + threadIdx.x;
  const int b = c / m;
  const float* cp = cpos + (size_t)c * 3;
  const float cx = cp[0], cy = cp[1], cz = cp[2];
  const float* pb = ppos + (size_t)b * Np * 3;
  const int base = threadIdx.x * 64;
  int cnt = 0, maxp = 0;
  float maxv = -1.0f;
  for (int p = 0; p < Np; ++p) {
    float dx = pb[p * 3] - cx, dy = pb[p * 3 + 1] - cy, dz = pb[p * 3 + 2] - cz;
    float d2 = dx * dx + dy * dy + dz * dz;
    if (d2 <= r2) {
      if (cnt < 64) {
        sd[base + cnt] = d2; si[base + cnt] = p;
        if (d2 > maxv) { maxv = d2; maxp = cnt; }
        ++cnt;
      } else if (d2 < maxv) {
        sd[base + maxp] = d2; si[base + maxp] = p;
        maxv = -1.0f;
        for (int k = 0; k < 64; ++k)
          if (sd[base + k] > maxv) { maxv = sd[base + k]; maxp = k; }
      }
    }
  }
  for (int k = 0; k < 64; ++k)
    nidx[(size_t)c * 64 + k] = (k < cnt) ? si[base + k] : -1;
}

// ---------------------------------------------------------------------------
// SA module 1, fully fused per group (64 neighbors):
//   feats[64x32] -> relu(W1a 32x64) -> relu(W1b 64x64) -> W1c 64x128
//   -> masked column max -> x1[group][128] (f16)
// Block = 128 threads (4 waves); wave w owns rows w*16..w*16+15.
// ---------------------------------------------------------------------------
__global__ __launch_bounds__(128) void k_sa1(
    const float* __restrict__ pos,   // [B*2048*3]
    const float* __restrict__ pos1,  // [B*1024*3] centroids
    const int* __restrict__ nidx,    // [B*1024*64]
    const _Float16* __restrict__ WT1a, const float* __restrict__ b1a,
    const _Float16* __restrict__ WT1b, const float* __restrict__ b1b,
    const _Float16* __restrict__ WT1c, const float* __restrict__ b1c,
    _Float16* __restrict__ x1)       // [B*1024*128]
{
  __shared__ _Float16 s_f[64 * 32];
  __shared__ _Float16 s_h1[64 * 64];
  __shared__ _Float16 s_h2[64 * 64];
  __shared__ float s_red[4 * 16];
  const int g = blockIdx.x, b = g >> 10, tid = threadIdx.x;
  const int w = tid >> 5, lane = tid & 31, ln = lane & 15, lh = lane >> 4;
  const float cx = pos1[g * 3], cy = pos1[g * 3 + 1], cz = pos1[g * 3 + 2];
  {  // build feats: rel pos (3) + zero pad to 32
    int row = tid >> 1, half = tid & 1;
    int n = nidx[(size_t)g * 64 + row];
    float rx = 0.f, ry = 0.f, rz = 0.f;
    if (n >= 0) {
      const float* p = pos + (size_t)(b * 2048 + n) * 3;
      rx = p[0] - cx; ry = p[1] - cy; rz = p[2] - cz;
    }
    _Float16* d = s_f + row * 32 + half * 16;
#pragma unroll
    for (int c = 0; c < 16; ++c) d[c] = (_Float16)0.0f;
    if (half == 0) { d[0] = (_Float16)rx; d[1] = (_Float16)ry; d[2] = (_Float16)rz; }
  }
  __syncthreads();
  {  // stage 1: K=32, N=64
    v16h a = load_a_frag(s_f + (w * 16) * 32, 32);
#pragma unroll
    for (int nt = 0; nt < 4; ++nt) {
      v8f acc = {};
      v16h bf = load_b_fragT(WT1a + (nt * 16) * 32, 32);
      acc = wmma_f16(a, bf, acc);
      float bias = b1a[nt * 16 + ln];
#pragma unroll
      for (int r = 0; r < 8; ++r) {
        float v = acc[r] + bias; v = v > 0.f ? v : 0.f;
        s_h1[(w * 16 + r + 8 * lh) * 64 + nt * 16 + ln] = (_Float16)v;
      }
    }
  }
  __syncthreads();
  {  // stage 2: K=64, N=64
    v16h a0 = load_a_frag(s_h1 + (w * 16) * 64, 64);
    v16h a1 = load_a_frag(s_h1 + (w * 16) * 64 + 32, 64);
#pragma unroll
    for (int nt = 0; nt < 4; ++nt) {
      v8f acc = {};
      acc = wmma_f16(a0, load_b_fragT(WT1b + (nt * 16) * 64, 64), acc);
      acc = wmma_f16(a1, load_b_fragT(WT1b + (nt * 16) * 64 + 32, 64), acc);
      float bias = b1b[nt * 16 + ln];
#pragma unroll
      for (int r = 0; r < 8; ++r) {
        float v = acc[r] + bias; v = v > 0.f ? v : 0.f;
        s_h2[(w * 16 + r + 8 * lh) * 64 + nt * 16 + ln] = (_Float16)v;
      }
    }
  }
  __syncthreads();
  // stage 3: K=64, N=128, masked max over the 64 rows of the group
  const bool nonempty = nidx[(size_t)g * 64] >= 0;
  bool valid[8];
#pragma unroll
  for (int r = 0; r < 8; ++r)
    valid[r] = nidx[(size_t)g * 64 + w * 16 + r + 8 * lh] >= 0;
  v16h a0 = load_a_frag(s_h2 + (w * 16) * 64, 64);
  v16h a1 = load_a_frag(s_h2 + (w * 16) * 64 + 32, 64);
  for (int nt = 0; nt < 8; ++nt) {
    v8f acc = {};
    acc = wmma_f16(a0, load_b_fragT(WT1c + (nt * 16) * 64, 64), acc);
    acc = wmma_f16(a1, load_b_fragT(WT1c + (nt * 16) * 64 + 32, 64), acc);
    float bias = b1c[nt * 16 + ln];
    float cmax = NEGINF;
#pragma unroll
    for (int r = 0; r < 8; ++r)
      if (valid[r]) cmax = fmaxf(cmax, acc[r] + bias);
    cmax = fmaxf(cmax, __shfl_xor(cmax, 16, 32));
    if (lane < 16) s_red[w * 16 + ln] = cmax;
    __syncthreads();
    if (tid < 16) {
      float mm = fmaxf(fmaxf(s_red[tid], s_red[16 + tid]),
                       fmaxf(s_red[32 + tid], s_red[48 + tid]));
      x1[(size_t)g * 128 + nt * 16 + tid] = (_Float16)(nonempty ? mm : 0.0f);
    }
    __syncthreads();
  }
}

// ---------------------------------------------------------------------------
// SA module 2, fused per group: feats[64x160] (x1 128 + rel 3 + pad)
//   -> relu(W2a 160x128) -> relu(W2b 128x128) -> W2c 128x256
//   -> masked column max -> x2[group][256]
// ---------------------------------------------------------------------------
__global__ __launch_bounds__(128) void k_sa2(
    const float* __restrict__ pos1,   // [B*1024*3] (neighbor source)
    const float* __restrict__ pos2,   // [B*256*3] centroids
    const int* __restrict__ nidx,     // [B*256*64] (indices into cloud's 1024)
    const _Float16* __restrict__ x1,  // [B*1024*128]
    const _Float16* __restrict__ WT2a, const float* __restrict__ b2a,
    const _Float16* __restrict__ WT2b, const float* __restrict__ b2b,
    const _Float16* __restrict__ WT2c, const float* __restrict__ b2c,
    _Float16* __restrict__ x2)        // [B*256*256]
{
  __shared__ _Float16 s_f[64 * 160];
  __shared__ _Float16 s_hA[64 * 128];
  __shared__ _Float16 s_hB[64 * 128];
  __shared__ float s_red[4 * 16];
  const int g = blockIdx.x, b = g >> 8, tid = threadIdx.x;
  const int w = tid >> 5, lane = tid & 31, ln = lane & 15, lh = lane >> 4;
  if (tid < 64) {  // one row per thread
    const int row = tid;
    const int n2 = nidx[(size_t)g * 64 + row];
    _Float16* d = s_f + row * 160;
    if (n2 >= 0) {
      const _Float16* xs = x1 + (size_t)(b * 1024 + n2) * 128;
      for (int c = 0; c < 128; ++c) d[c] = xs[c];
      const float* p = pos1 + (size_t)(b * 1024 + n2) * 3;
      d[128] = (_Float16)(p[0] - pos2[g * 3]);
      d[129] = (_Float16)(p[1] - pos2[g * 3 + 1]);
      d[130] = (_Float16)(p[2] - pos2[g * 3 + 2]);
      for (int c = 131; c < 160; ++c) d[c] = (_Float16)0.0f;
    } else {
      for (int c = 0; c < 160; ++c) d[c] = (_Float16)0.0f;
    }
  }
  __syncthreads();
  {  // stage A: K=160 (5 steps), N=128
    v16h af[5];
#pragma unroll
    for (int ks = 0; ks < 5; ++ks)
      af[ks] = load_a_frag(s_f + (w * 16) * 160 + ks * 32, 160);
    for (int nt = 0; nt < 8; ++nt) {
      v8f acc = {};
#pragma unroll
      for (int ks = 0; ks < 5; ++ks)
        acc = wmma_f16(af[ks], load_b_fragT(WT2a + (nt * 16) * 160 + ks * 32, 160), acc);
      float bias = b2a[nt * 16 + ln];
#pragma unroll
      for (int r = 0; r < 8; ++r) {
        float v = acc[r] + bias; v = v > 0.f ? v : 0.f;
        s_hA[(w * 16 + r + 8 * lh) * 128 + nt * 16 + ln] = (_Float16)v;
      }
    }
  }
  __syncthreads();
  {  // stage B: K=128 (4 steps), N=128
    v16h af[4];
#pragma unroll
    for (int ks = 0; ks < 4; ++ks)
      af[ks] = load_a_frag(s_hA + (w * 16) * 128 + ks * 32, 128);
    for (int nt = 0; nt < 8; ++nt) {
      v8f acc = {};
#pragma unroll
      for (int ks = 0; ks < 4; ++ks)
        acc = wmma_f16(af[ks], load_b_fragT(WT2b + (nt * 16) * 128 + ks * 32, 128), acc);
      float bias = b2b[nt * 16 + ln];
#pragma unroll
      for (int r = 0; r < 8; ++r) {
        float v = acc[r] + bias; v = v > 0.f ? v : 0.f;
        s_hB[(w * 16 + r + 8 * lh) * 128 + nt * 16 + ln] = (_Float16)v;
      }
    }
  }
  __syncthreads();
  // stage C: K=128, N=256, masked max
  const bool nonempty = nidx[(size_t)g * 64] >= 0;
  bool valid[8];
#pragma unroll
  for (int r = 0; r < 8; ++r)
    valid[r] = nidx[(size_t)g * 64 + w * 16 + r + 8 * lh] >= 0;
  v16h af[4];
#pragma unroll
  for (int ks = 0; ks < 4; ++ks)
    af[ks] = load_a_frag(s_hB + (w * 16) * 128 + ks * 32, 128);
  for (int nt = 0; nt < 16; ++nt) {
    v8f acc = {};
#pragma unroll
    for (int ks = 0; ks < 4; ++ks)
      acc = wmma_f16(af[ks], load_b_fragT(WT2c + (nt * 16) * 128 + ks * 32, 128), acc);
    float bias = b2c[nt * 16 + ln];
    float cmax = NEGINF;
#pragma unroll
    for (int r = 0; r < 8; ++r)
      if (valid[r]) cmax = fmaxf(cmax, acc[r] + bias);
    cmax = fmaxf(cmax, __shfl_xor(cmax, 16, 32));
    if (lane < 16) s_red[w * 16 + ln] = cmax;
    __syncthreads();
    if (tid < 16) {
      float mm = fmaxf(fmaxf(s_red[tid], s_red[16 + tid]),
                       fmaxf(s_red[32 + tid], s_red[48 + tid]));
      x2[(size_t)g * 256 + nt * 16 + tid] = (_Float16)(nonempty ? mm : 0.0f);
    }
    __syncthreads();
  }
}

// ---------------------------------------------------------------------------
// feats3 = concat(x2, pos2) padded 259 -> 288, one row per thread
// ---------------------------------------------------------------------------
__global__ __launch_bounds__(256) void k_feats3(const _Float16* __restrict__ x2,
                                                const float* __restrict__ pos2,
                                                _Float16* __restrict__ f3) {
  int row = blockIdx.x * 256 + threadIdx.x;  // 0..2047
  _Float16* d = f3 + (size_t)row * 288;
  const _Float16* xs = x2 + (size_t)row * 256;
  for (int c = 0; c < 256; ++c) d[c] = xs[c];
  d[256] = (_Float16)pos2[row * 3];
  d[257] = (_Float16)pos2[row * 3 + 1];
  d[258] = (_Float16)pos2[row * 3 + 2];
  for (int c = 259; c < 288; ++c) d[c] = (_Float16)0.0f;
}

// ---------------------------------------------------------------------------
// Generic WMMA GEMM: out = act(A[M x lda] @ WT^T + bias), f16 out.
// Block 128 = 4 waves, each wave one 16-row subtile of a 64-row block tile.
// ---------------------------------------------------------------------------
__global__ __launch_bounds__(128) void k_gemm16(const _Float16* __restrict__ A,
                                                int lda,
                                                const _Float16* __restrict__ WT,
                                                int Kp,
                                                const float* __restrict__ bias,
                                                _Float16* __restrict__ out,
                                                int ldo, int relu) {
  const int w = threadIdx.x >> 5, lane = threadIdx.x & 31;
  const int ln = lane & 15, lh = lane >> 4;
  const int row0 = blockIdx.x * 64 + w * 16;
  const int n0 = blockIdx.y * 16;
  v8f acc = {};
  for (int k0 = 0; k0 < Kp; k0 += 32) {
    __builtin_prefetch(A + (size_t)row0 * lda + k0 + 32, 0, 0);
    v16h a = load_a_frag(A + (size_t)row0 * lda + k0, lda);
    v16h bf = load_b_fragT(WT + (size_t)n0 * Kp + k0, Kp);
    acc = wmma_f16(a, bf, acc);
  }
  const float bb = bias[n0 + ln];
#pragma unroll
  for (int r = 0; r < 8; ++r) {
    float v = acc[r] + bb;
    if (relu) v = v > 0.f ? v : 0.f;
    out[(size_t)(row0 + r + 8 * lh) * ldo + n0 + ln] = (_Float16)v;
  }
}

// ---------------------------------------------------------------------------
// Final MLP3 layer (512 -> 1024) fused with per-cloud global max pool.
// grid = (B, 1024/16); block loops over the cloud's 4 x 64-row chunks.
// ---------------------------------------------------------------------------
__global__ __launch_bounds__(128) void k_gemm_gmax(
    const _Float16* __restrict__ A,   // [B*256 x 512]
    const _Float16* __restrict__ WT,  // [1024 x 512]
    const float* __restrict__ bias, float* __restrict__ gfeat) {
  __shared__ float s_red[4 * 16];
  const int w = threadIdx.x >> 5, lane = threadIdx.x & 31;
  const int ln = lane & 15, lh = lane >> 4;
  (void)lh;
  const int b = blockIdx.x;
  const int n0 = blockIdx.y * 16;
  float cmax = NEGINF;
  for (int chunk = 0; chunk < 4; ++chunk) {
    const int row0 = b * 256 + chunk * 64 + w * 16;
    v8f acc = {};
    for (int k0 = 0; k0 < 512; k0 += 32) {
      __builtin_prefetch(A + (size_t)row0 * 512 + k0 + 32, 0, 0);
      v16h a = load_a_frag(A + (size_t)row0 * 512 + k0, 512);
      v16h bf = load_b_fragT(WT + (size_t)n0 * 512 + k0, 512);
      acc = wmma_f16(a, bf, acc);
    }
#pragma unroll
    for (int r = 0; r < 8; ++r) cmax = fmaxf(cmax, acc[r]);
  }
  cmax = fmaxf(cmax, __shfl_xor(cmax, 16, 32));
  if (lane < 16) s_red[w * 16 + ln] = cmax;
  __syncthreads();
  if (threadIdx.x < 16) {
    float mm = fmaxf(fmaxf(s_red[threadIdx.x], s_red[16 + threadIdx.x]),
                     fmaxf(s_red[32 + threadIdx.x], s_red[48 + threadIdx.x]));
    gfeat[b * 1024 + n0 + threadIdx.x] = mm + bias[n0 + threadIdx.x];
  }
}

// ---------------------------------------------------------------------------
// Classifier head (8 rows): 1024 -> 512 -> 256 -> 10, log_softmax. f32.
// ---------------------------------------------------------------------------
__global__ __launch_bounds__(256) void k_head(const float* __restrict__ gfeat,
                                              const float* __restrict__ Wh1,
                                              const float* __restrict__ bh1,
                                              const float* __restrict__ Wh2,
                                              const float* __restrict__ bh2,
                                              const float* __restrict__ Wh3,
                                              const float* __restrict__ bh3,
                                              float* __restrict__ out) {
  __shared__ float s1[512];
  __shared__ float s2[256];
  __shared__ float sl[10];
  __shared__ float slse;
  const int b = blockIdx.x, tid = threadIdx.x;
  const float* g = gfeat + b * 1024;
  for (int o = tid; o < 512; o += 256) {
    float acc = bh1[o];
    for (int i = 0; i < 1024; ++i) acc += g[i] * Wh1[(size_t)i * 512 + o];
    s1[o] = acc > 0.f ? acc : 0.f;
  }
  __syncthreads();
  {
    float acc = bh2[tid];
    for (int i = 0; i < 512; ++i) acc += s1[i] * Wh2[(size_t)i * 256 + tid];
    s2[tid] = acc > 0.f ? acc : 0.f;
  }
  __syncthreads();
  if (tid < 10) {
    float acc = bh3[tid];
    for (int i = 0; i < 256; ++i) acc += s2[i] * Wh3[i * 10 + tid];
    sl[tid] = acc;
  }
  __syncthreads();
  if (tid == 0) {
    float mx = sl[0];
    for (int j = 1; j < 10; ++j) mx = fmaxf(mx, sl[j]);
    float s = 0.f;
    for (int j = 0; j < 10; ++j) s += expf(sl[j] - mx);
    slse = mx + logf(s);
  }
  __syncthreads();
  if (tid < 10) out[b * 10 + tid] = sl[tid] - slse;
}

// ---------------------------------------------------------------------------
// Host side
// ---------------------------------------------------------------------------
extern "C" void kernel_launch(void* const* d_in, const int* in_sizes, int n_in,
                              void* d_out, int out_size, void* d_ws,
                              size_t ws_size, hipStream_t stream) {
  (void)in_sizes; (void)n_in; (void)out_size; (void)ws_size;
  const float* pos = (const float*)d_in[0];
  const float* W1a = (const float*)d_in[1];  const float* b1a = (const float*)d_in[2];
  const float* W1b = (const float*)d_in[3];  const float* b1b = (const float*)d_in[4];
  const float* W1c = (const float*)d_in[5];  const float* b1c = (const float*)d_in[6];
  const float* W2a = (const float*)d_in[7];  const float* b2a = (const float*)d_in[8];
  const float* W2b = (const float*)d_in[9];  const float* b2b = (const float*)d_in[10];
  const float* W2c = (const float*)d_in[11]; const float* b2c = (const float*)d_in[12];
  const float* W3a = (const float*)d_in[13]; const float* b3a = (const float*)d_in[14];
  const float* W3b = (const float*)d_in[15]; const float* b3b = (const float*)d_in[16];
  const float* W3c = (const float*)d_in[17]; const float* b3c = (const float*)d_in[18];
  const float* Wh1 = (const float*)d_in[19]; const float* bh1 = (const float*)d_in[20];
  const float* Wh2 = (const float*)d_in[21]; const float* bh2 = (const float*)d_in[22];
  const float* Wh3 = (const float*)d_in[23]; const float* bh3 = (const float*)d_in[24];

  // bump allocator over d_ws (deterministic layout, ~12 MB total)
  size_t off = 0;
  char* base = (char*)d_ws;
  auto alloc = [&](size_t bytes) -> void* {
    void* p = base + off;
    off = (off + bytes + 255) & ~(size_t)255;
    return p;
  };
  float* curv  = (float*)alloc(8 * 2048 * 4);
  int* idx1    = (int*)alloc(8 * 1024 * 4);
  float* pos1  = (float*)alloc(8 * 1024 * 3 * 4);
  float* curv1 = (float*)alloc(8 * 1024 * 4);
  int* idx2    = (int*)alloc(8 * 256 * 4);
  float* pos2  = (float*)alloc(8 * 256 * 3 * 4);
  int* nidx1   = (int*)alloc((size_t)8 * 1024 * 64 * 4);
  int* nidx2   = (int*)alloc((size_t)8 * 256 * 64 * 4);
  _Float16* x1 = (_Float16*)alloc((size_t)8 * 1024 * 128 * 2);
  _Float16* x2 = (_Float16*)alloc((size_t)8 * 256 * 256 * 2);
  _Float16* f3 = (_Float16*)alloc((size_t)2048 * 288 * 2);
  _Float16* h3a = (_Float16*)alloc((size_t)2048 * 256 * 2);
  _Float16* h3b = (_Float16*)alloc((size_t)2048 * 512 * 2);
  float* gfeat = (float*)alloc(8 * 1024 * 4);
  _Float16* WT1a = (_Float16*)alloc(64 * 32 * 2);
  _Float16* WT1b = (_Float16*)alloc(64 * 64 * 2);
  _Float16* WT1c = (_Float16*)alloc(128 * 64 * 2);
  _Float16* WT2a = (_Float16*)alloc(128 * 160 * 2);
  _Float16* WT2b = (_Float16*)alloc(128 * 128 * 2);
  _Float16* WT2c = (_Float16*)alloc(256 * 128 * 2);
  _Float16* WT3a = (_Float16*)alloc(256 * 288 * 2);
  _Float16* WT3b = (_Float16*)alloc(512 * 256 * 2);
  _Float16* WT3c = (_Float16*)alloc((size_t)1024 * 512 * 2);

  auto pack = [&](const float* W, _Float16* WT, int K, int N, int Kp) {
    k_packWT<<<(N * Kp + 255) / 256, 256, 0, stream>>>(W, WT, K, N, Kp);
  };
  pack(W1a, WT1a, 3, 64, 32);
  pack(W1b, WT1b, 64, 64, 64);
  pack(W1c, WT1c, 64, 128, 64);
  pack(W2a, WT2a, 131, 128, 160);
  pack(W2b, WT2b, 128, 128, 128);
  pack(W2c, WT2c, 128, 256, 128);
  pack(W3a, WT3a, 259, 256, 288);
  pack(W3b, WT3b, 256, 512, 256);
  pack(W3c, WT3c, 512, 1024, 512);

  // curvature + level-1 sampling/grouping
  k_curv<<<dim3(8, 8), 256, 0, stream>>>(pos, curv);
  k_fps<<<8, 256, 0, stream>>>(pos, curv, 2048, 1024, idx1);
  k_gather<<<(8192 + 255) / 256, 256, 0, stream>>>(pos, curv, idx1, 2048, 1024,
                                                   8192, pos1, curv1);
  k_group<<<128, 64, 0, stream>>>(pos1, pos, 2048, 1024, 0.2f * 0.2f, nidx1);
  k_sa1<<<8192, 128, 0, stream>>>(pos, pos1, nidx1, WT1a, b1a, WT1b, b1b, WT1c,
                                  b1c, x1);

  // level-2 sampling/grouping
  k_fps<<<8, 256, 0, stream>>>(pos1, curv1, 1024, 256, idx2);
  k_gather<<<(2048 + 255) / 256, 256, 0, stream>>>(pos1, nullptr, idx2, 1024,
                                                   256, 2048, pos2, nullptr);
  k_group<<<32, 64, 0, stream>>>(pos2, pos1, 1024, 256, 0.4f * 0.4f, nidx2);
  k_sa2<<<2048, 128, 0, stream>>>(pos1, pos2, nidx2, x1, WT2a, b2a, WT2b, b2b,
                                  WT2c, b2c, x2);

  // global MLP3 + max pool
  k_feats3<<<8, 256, 0, stream>>>(x2, pos2, f3);
  k_gemm16<<<dim3(32, 16), 128, 0, stream>>>(f3, 288, WT3a, 288, b3a, h3a, 256, 1);
  k_gemm16<<<dim3(32, 32), 128, 0, stream>>>(h3a, 256, WT3b, 256, b3b, h3b, 512, 1);
  k_gemm_gmax<<<dim3(8, 64), 128, 0, stream>>>(h3b, WT3c, b3c, gfeat);

  // classifier head + log_softmax
  k_head<<<8, 256, 0, stream>>>(gfeat, Wh1, bh1, Wh2, bh2, Wh3, bh3,
                                (float*)d_out);
}
